// LSTM_59322088292480
// MI455X (gfx1250) — matrix operand order repaired and checked by reading, not compiled, not run on hardware
//
#include <hip/hip_runtime.h>

typedef __attribute__((ext_vector_type(16))) _Float16 v16h;
typedef __attribute__((ext_vector_type(8)))  float    v8f;

#define BB 1024
#define SS 512
#define FF 18
#define HH 64
#define OO 15

union Frag { v16h v; uint4 q[2]; };

// A-matrix f16 16x32 fragment from row-major [16][ld] LDS tile.
// Lane L(0..15): row M=L, K = {k0..k0+7} U {k0+16..k0+23}; lane L+16: +8 on K base.
__device__ __forceinline__ v16h ldA(const _Float16* base, int ld, int lane, int k0) {
  int row = lane & 15;
  int kb  = k0 + ((lane >> 4) << 3);
  const _Float16* p = base + row * ld + kb;
  Frag f;
  f.q[0] = *reinterpret_cast<const uint4*>(p);
  f.q[1] = *reinterpret_cast<const uint4*>(p + 16);
  return f.v;
}

// B-matrix f16 32x16 fragment: B[k][n] = W[ntile*16 + n][k] with W row-major [256][ldw].
// Lane L: column N=L&15, K = {kb..kb+7} U {kb+16..kb+23}, kb = k0 + (L>=16?8:0).
__device__ __forceinline__ v16h ldB(const _Float16* W, int ldw, int lane, int ntile, int k0) {
  int n  = ntile * 16 + (lane & 15);
  int kb = k0 + ((lane >> 4) << 3);
  const _Float16* p = W + n * ldw + kb;
  Frag f;
  f.q[0] = *reinterpret_cast<const uint4*>(p);
  f.q[1] = *reinterpret_cast<const uint4*>(p + 16);
  return f.v;
}

__device__ __forceinline__ v8f wmma16(v16h a, v16h b, v8f c) {
  return __builtin_amdgcn_wmma_f32_16x16x32_f16(false, a, false, b, (short)0, c, false, false);
}

__device__ __forceinline__ float sigmoidf_(float x) { return 1.f / (1.f + __expf(-x)); }
__device__ __forceinline__ float tanhf_(float x)    { return 1.f - 2.f / (__expf(2.f * x) + 1.f); }

__global__ void __launch_bounds__(128) lstm_fused(
    const float* __restrict__ X,
    const float* __restrict__ Wih0, const float* __restrict__ Whh0,
    const float* __restrict__ bih0, const float* __restrict__ bhh0,
    const float* __restrict__ Wih1, const float* __restrict__ Whh1,
    const float* __restrict__ bih1, const float* __restrict__ bhh1,
    float* __restrict__ lastH)
{
  extern __shared__ char smem[];
  _Float16* sWih0 = (_Float16*)smem;        // [256][32] (K padded 18->32 with zeros)
  _Float16* sWhh0 = sWih0 + 256 * 32;       // [256][64]
  _Float16* sWih1 = sWhh0 + 256 * 64;       // [256][64]
  _Float16* sWhh1 = sWih1 + 256 * 64;       // [256][64]
  _Float16* sX    = sWhh1 + 256 * 64;       // [16][32]  (F padded 18->32)
  _Float16* sHA   = sX    + 16 * 32;        // [16][64]  layer-0 hidden
  _Float16* sHB   = sHA   + 16 * 64;        // [16][64]  layer-1 hidden

  const int tid  = threadIdx.x;
  const int lane = tid & 31;
  const int w    = tid >> 5;                // wave id 0..3 -> hidden-unit tile
  const int bb0  = blockIdx.x * 16;         // batch tile

  // ---- stage weights to LDS as f16 (once) ----
  #pragma unroll 4
  for (int idx = tid; idx < 256 * 32; idx += 128) {
    int n = idx >> 5, k = idx & 31;
    sWih0[idx] = (k < FF) ? (_Float16)Wih0[n * FF + k] : (_Float16)0.f;
  }
  #pragma unroll 4
  for (int idx = tid; idx < 256 * 64; idx += 128) sWhh0[idx] = (_Float16)Whh0[idx];
  #pragma unroll 4
  for (int idx = tid; idx < 256 * 64; idx += 128) sWih1[idx] = (_Float16)Wih1[idx];
  #pragma unroll 4
  for (int idx = tid; idx < 256 * 64; idx += 128) sWhh1[idx] = (_Float16)Whh1[idx];
  #pragma unroll
  for (int idx = tid; idx < 16 * 64; idx += 128) { sHA[idx] = (_Float16)0.f; sHB[idx] = (_Float16)0.f; }

  // ---- per-lane gate biases (column = g*64 + w*16 + lane%16) ----
  const int nloc = lane & 15;
  float bias0[4], bias1[4];
  #pragma unroll
  for (int g = 0; g < 4; ++g) {
    int n = g * 64 + w * 16 + nloc;
    bias0[g] = bih0[n] + bhh0[n];
    bias1[g] = bih1[n] + bhh1[n];
  }

  v8f c0 = {}, c1 = {};
  __syncthreads();

  for (int s = 0; s < SS; ++s) {
    // stage X[:,s,:] tile (f32 -> f16, padded)
    #pragma unroll
    for (int idx = tid; idx < 16 * 32; idx += 128) {
      int r = idx >> 5, k = idx & 31;
      sX[idx] = (k < FF) ? (_Float16)X[((size_t)(bb0 + r) * SS + s) * FF + k] : (_Float16)0.f;
    }
    __syncthreads();

    v16h ax   = ldA(sX,  32, lane, 0);
    v16h ahA0 = ldA(sHA, 64, lane, 0);
    v16h ahA1 = ldA(sHA, 64, lane, 32);
    v16h ahB0 = ldA(sHB, 64, lane, 0);   // old layer-1 state (read before overwrite)
    v16h ahB1 = ldA(sHB, 64, lane, 32);

    // ---- layer 0 gates: x @ Wih0^T + hA @ Whh0^T ----
    v8f acc[4];
    #pragma unroll
    for (int g = 0; g < 4; ++g) {
      int t = g * 4 + w;
      v8f a = {};
      a = wmma16(ax,   ldB(sWih0, 32, lane, t, 0),  a);
      a = wmma16(ahA0, ldB(sWhh0, 64, lane, t, 0),  a);
      a = wmma16(ahA1, ldB(sWhh0, 64, lane, t, 32), a);
      acc[g] = a;
    }
    float hAn[8];
    #pragma unroll
    for (int i = 0; i < 8; ++i) {
      float ig = sigmoidf_(acc[0][i] + bias0[0]);
      float fg = sigmoidf_(acc[1][i] + bias0[1]);
      float gg = tanhf_   (acc[2][i] + bias0[2]);
      float og = sigmoidf_(acc[3][i] + bias0[3]);
      float c  = fg * c0[i] + ig * gg;
      c0[i] = c;
      hAn[i] = og * tanhf_(c);
    }
    __syncthreads();   // all waves finished reading old sHA
    {
      int mb = (lane >> 4) << 3;
      #pragma unroll
      for (int i = 0; i < 8; ++i)
        sHA[(mb + i) * 64 + w * 16 + nloc] = (_Float16)hAn[i];
    }
    __syncthreads();   // new sHA visible to all waves

    v16h ahN0 = ldA(sHA, 64, lane, 0);
    v16h ahN1 = ldA(sHA, 64, lane, 32);

    // ---- layer 1 gates: hA_new @ Wih1^T + hB_old @ Whh1^T ----
    #pragma unroll
    for (int g = 0; g < 4; ++g) {
      int t = g * 4 + w;
      v8f a = {};
      a = wmma16(ahN0, ldB(sWih1, 64, lane, t, 0),  a);
      a = wmma16(ahN1, ldB(sWih1, 64, lane, t, 32), a);
      a = wmma16(ahB0, ldB(sWhh1, 64, lane, t, 0),  a);
      a = wmma16(ahB1, ldB(sWhh1, 64, lane, t, 32), a);
      acc[g] = a;
    }
    float hBn[8];
    #pragma unroll
    for (int i = 0; i < 8; ++i) {
      float ig = sigmoidf_(acc[0][i] + bias1[0]);
      float fg = sigmoidf_(acc[1][i] + bias1[1]);
      float gg = tanhf_   (acc[2][i] + bias1[2]);
      float og = sigmoidf_(acc[3][i] + bias1[3]);
      float c  = fg * c1[i] + ig * gg;
      c1[i] = c;
      hBn[i] = og * tanhf_(c);
    }
    __syncthreads();   // all waves finished reading old sHB
    {
      int mb = (lane >> 4) << 3;
      #pragma unroll
      for (int i = 0; i < 8; ++i)
        sHB[(mb + i) * 64 + w * 16 + nloc] = (_Float16)hBn[i];
    }
    __syncthreads();

    if (s == SS - 1) {
      int mb = (lane >> 4) << 3;
      #pragma unroll
      for (int i = 0; i < 8; ++i)
        lastH[(size_t)(bb0 + mb + i) * 64 + w * 16 + nloc] = hBn[i];
    }
  }
}

// Batch-norm statistics over batch dim of last hidden state: per column j,
// store a[j] = gamma*rsqrt(var+eps) and b2[j] = beta - mean*a.
__global__ void bn_stats(const float* __restrict__ lastH,
                         const float* __restrict__ gamma, const float* __restrict__ beta,
                         float* __restrict__ ab)
{
  int j = threadIdx.x;  // 64 threads
  float sum = 0.f, sq = 0.f;
  #pragma unroll 8
  for (int b = 0; b < BB; ++b) {
    float x = lastH[b * HH + j];
    sum += x; sq += x * x;
  }
  float mean = sum * (1.f / BB);
  float var  = sq * (1.f / BB) - mean * mean;
  float a = gamma[j] * rsqrtf(var + 1e-5f);
  ab[j]      = a;
  ab[HH + j] = beta[j] - mean * a;
}

__global__ void classify(const float* __restrict__ lastH, const float* __restrict__ ab,
                         const float* __restrict__ clsW, const float* __restrict__ clsB,
                         float* __restrict__ out)
{
  int b = blockIdx.x;
  int o = threadIdx.x;
  if (o < OO) {
    float acc = clsB[o];
    #pragma unroll 8
    for (int j = 0; j < HH; ++j)
      acc += (lastH[b * HH + j] * ab[j] + ab[HH + j]) * clsW[o * HH + j];
    out[b * OO + o] = acc;
  }
}

extern "C" void kernel_launch(void* const* d_in, const int* in_sizes, int n_in,
                              void* d_out, int out_size, void* d_ws, size_t ws_size,
                              hipStream_t stream) {
  const float* X    = (const float*)d_in[0];
  const float* Wih0 = (const float*)d_in[1];
  const float* Whh0 = (const float*)d_in[2];
  const float* bih0 = (const float*)d_in[3];
  const float* bhh0 = (const float*)d_in[4];
  const float* Wih1 = (const float*)d_in[5];
  const float* Whh1 = (const float*)d_in[6];
  const float* bih1 = (const float*)d_in[7];
  const float* bhh1 = (const float*)d_in[8];
  const float* gam  = (const float*)d_in[9];
  const float* bet  = (const float*)d_in[10];
  const float* clsW = (const float*)d_in[11];
  const float* clsB = (const float*)d_in[12];
  float* out   = (float*)d_out;
  float* lastH = (float*)d_ws;              // 1024*64 f32 = 256 KB
  float* ab    = lastH + BB * HH;           // 128 f32

  size_t smem = (size_t)(256 * 32 + 3 * 256 * 64 + 16 * 32 + 2 * 16 * 64) * sizeof(_Float16);
  lstm_fused<<<BB / 16, 128, smem, stream>>>(X, Wih0, Whh0, bih0, bhh0,
                                             Wih1, Whh1, bih1, bhh1, lastH);
  bn_stats<<<1, HH, 0, stream>>>(lastH, gam, bet, ab);
  classify<<<BB, 32, 0, stream>>>(lastH, ab, clsW, clsB, out);
}
